// PointCloudTokenizer_66855460930294
// MI455X (gfx1250) — matrix operand order repaired and checked by reading, not compiled, not run on hardware
//
#include <hip/hip_runtime.h>
#include <hip/hip_bf16.h>

// ---------------------------------------------------------------------------
// PointCloudTokenizer for MI455X (gfx1250, wave32, WMMA)
//
// Pipeline:
//   1) wcvt:       transpose+convert weights f32[K,N] -> f16[N,Kpad] (W1 padded K=6->32)
//   2) fps_kernel: farthest point sampling per event (16 blocks x 1024 thr)
//   3) knn_kernel: top-16 nearest points per centroid (2048 blocks)
//   4) token_kernel: fused 4-layer point-MLP on gathered neighbors using
//      v_wmma_f32_16x16x32_f16. 4 tokens per block (M=64 rows) to amortize
//      weight traffic 4x (L2-bound -> compute-bound). Activations ping-pong in
//      192KB of LDS (ISA: up to 320KB/workgroup). Max-pool -> pooled f16.
//   5) gemm_tok x2: attention MLP (2048x768 @ 768x768) with WMMA
//   6) fill masks = 1.0f
// ---------------------------------------------------------------------------

#define B_EV 16
#define P_PT 8192
#define FEAT 6
#define MTOK 128
#define TD   768
#define KNN  16
#define TPB_TOK 4              // tokens per block in token_kernel

typedef __attribute__((ext_vector_type(16))) _Float16 v16h;
typedef __attribute__((ext_vector_type(8)))  _Float16 v8h;
typedef __attribute__((ext_vector_type(8)))  float    v8f;

// ---------------- weight transpose/convert: W[K,N] f32 -> Wt[N,Kpad] f16 ----
__global__ void wcvt_kernel(const float* __restrict__ W, _Float16* __restrict__ Wt,
                            int K, int N, int Kpad) {
  int id = blockIdx.x * 256 + threadIdx.x;
  int total = N * Kpad;
  if (id >= total) return;
  int n = id / Kpad;
  int k = id - n * Kpad;
  float v = (k < K) ? W[(size_t)k * N + n] : 0.0f;
  Wt[id] = (_Float16)v;
}

// ---------------- farthest point sampling ----------------------------------
__global__ __launch_bounds__(1024) void fps_kernel(const float* __restrict__ coords,
                                                   float* __restrict__ centOut,
                                                   float* __restrict__ centWs,
                                                   int* __restrict__ fpsIdx) {
  __shared__ float sVal[1024];
  __shared__ int   sIdx[1024];
  __shared__ float cur[4];
  const int tid = threadIdx.x;
  const int b   = blockIdx.x;
  float px[8], py[8], pz[8], pw[8], dmin[8];
#pragma unroll
  for (int i = 0; i < 8; ++i) {
    int p = tid + i * 1024;
    const float* c = coords + ((size_t)(b * P_PT + p)) * 5 + 1;
    px[i] = c[0]; py[i] = c[1]; pz[i] = c[2]; pw[i] = c[3];
    dmin[i] = 1e30f;
  }
  if (tid == 0) {
    cur[0] = px[0]; cur[1] = py[0]; cur[2] = pz[0]; cur[3] = pw[0];
    fpsIdx[b * MTOK] = 0;
    float* co = centOut + ((size_t)b * MTOK) * 4;
    co[0] = px[0]; co[1] = py[0]; co[2] = pz[0]; co[3] = pw[0];
    float* cw = centWs + ((size_t)b * MTOK) * 4;
    cw[0] = px[0]; cw[1] = py[0]; cw[2] = pz[0]; cw[3] = pw[0];
  }
  __syncthreads();
  for (int j = 1; j < MTOK; ++j) {
    const float c0 = cur[0], c1 = cur[1], c2 = cur[2], c3 = cur[3];
    float bv = -1.0f; int bi = 0;
#pragma unroll
    for (int i = 0; i < 8; ++i) {
      float dx = px[i] - c0, dy = py[i] - c1, dz = pz[i] - c2, dw = pw[i] - c3;
      float d = dx * dx + dy * dy + dz * dz + dw * dw;
      dmin[i] = fminf(dmin[i], d);
      int p = tid + i * 1024;
      if (dmin[i] > bv || (dmin[i] == bv && p < bi)) { bv = dmin[i]; bi = p; }
    }
    sVal[tid] = bv; sIdx[tid] = bi;
    __syncthreads();
    for (int s = 512; s > 0; s >>= 1) {
      if (tid < s) {
        float ov = sVal[tid + s]; int oi = sIdx[tid + s];
        if (ov > sVal[tid] || (ov == sVal[tid] && oi < sIdx[tid])) {
          sVal[tid] = ov; sIdx[tid] = oi;
        }
      }
      __syncthreads();
    }
    int nxt = sIdx[0];
    if (tid == (nxt & 1023)) {
      int slot = nxt >> 10;
      cur[0] = px[slot]; cur[1] = py[slot]; cur[2] = pz[slot]; cur[3] = pw[slot];
    }
    __syncthreads();
    if (tid == 0) {
      fpsIdx[b * MTOK + j] = nxt;
      float* co = centOut + ((size_t)(b * MTOK + j)) * 4;
      co[0] = cur[0]; co[1] = cur[1]; co[2] = cur[2]; co[3] = cur[3];
      float* cw = centWs + ((size_t)(b * MTOK + j)) * 4;
      cw[0] = cur[0]; cw[1] = cur[1]; cw[2] = cur[2]; cw[3] = cur[3];
    }
  }
}

// ---------------- kNN: 16 smallest of 8192 per centroid ---------------------
__global__ __launch_bounds__(256) void knn_kernel(const float* __restrict__ coords,
                                                  const float* __restrict__ centWs,
                                                  int* __restrict__ knnIdx) {
  __shared__ float sd[P_PT];
  __shared__ float rv[256];
  __shared__ int   ri[256];
  const int tid = threadIdx.x;
  const int tok = blockIdx.x;       // b*128 + m
  const int b   = tok >> 7;
  const float c0 = centWs[tok * 4 + 0], c1 = centWs[tok * 4 + 1];
  const float c2 = centWs[tok * 4 + 2], c3 = centWs[tok * 4 + 3];
  for (int i = 0; i < 32; ++i) {
    int p = tid + i * 256;
    const float* c = coords + ((size_t)(b * P_PT + p)) * 5 + 1;
    float dx = c[0] - c0, dy = c[1] - c1, dz = c[2] - c2, dw = c[3] - c3;
    sd[p] = dx * dx + dy * dy + dz * dz + dw * dw;
  }
  __syncthreads();
  for (int r = 0; r < KNN; ++r) {
    float bv = 1e31f; int bi = 0;
    for (int i = 0; i < 32; ++i) {
      int p = tid + i * 256;
      float d = sd[p];
      if (d < bv || (d == bv && p < bi)) { bv = d; bi = p; }
    }
    rv[tid] = bv; ri[tid] = bi;
    __syncthreads();
    for (int s = 128; s > 0; s >>= 1) {
      if (tid < s) {
        float ov = rv[tid + s]; int oi = ri[tid + s];
        if (ov < rv[tid] || (ov == rv[tid] && oi < ri[tid])) {
          rv[tid] = ov; ri[tid] = oi;
        }
      }
      __syncthreads();
    }
    if (tid == 0) {
      int w = ri[0];
      knnIdx[tok * KNN + r] = w;
      sd[w] = 1e30f;          // remove from candidate set
    }
    __syncthreads();
  }
}

// ---------------- one MLP layer tile-worker (M=64 block tile) via WMMA ------
// A fragment (16-bit A 16x32): lanes 0-15 -> row=lane, K={0..7,16..23};
//                              lanes 16-31 -> row=lane-16, K={8..15,24..31}.
// B fragment (16-bit B 32x16): lanes 0-15 -> col=lane, K={0..7,16..23};
//                              lanes 16-31 -> col=lane-16, K={8..15,24..31}.
// C/D: VGPR r -> m = r + 8*(lane>=16), n = n0 + (lane & 15).
// Wave handles row-tile mt (rows mt*16..mt*16+15) and N-tiles nt=ng,ng+NG,...
__device__ __forceinline__ void mlp_layer4(const _Float16* sIn, int Kdim,
                                           const _Float16* __restrict__ Wt,
                                           const float* __restrict__ bias,
                                           _Float16* sOut, int Ndim, bool relu,
                                           int mt, int ng, int ngroups, int lane) {
  const int half = lane >> 4;
  const int l15  = lane & 15;
  const int sa   = half ? 8 : 0;    // A lane-half K offset
  const int sb   = half ? 8 : 0;    // B lane-half K offset
  const int row0 = mt << 4;
  const _Float16* aRow = sIn + (row0 + l15) * Kdim;
  const int ntiles = Ndim >> 4;
  for (int nt = ng; nt < ntiles; nt += ngroups) {
    const int n0 = nt << 4;
    const _Float16* bRow = Wt + (size_t)(n0 + l15) * Kdim;
    v8f acc = {};
    for (int k0 = 0; k0 < Kdim; k0 += 32) {
      union { v16h v; v8h h[2]; } a, b;
      a.h[0] = *(const v8h*)(aRow + k0 + sa);          // ds_load_b128
      a.h[1] = *(const v8h*)(aRow + k0 + sa + 16);
      b.h[0] = *(const v8h*)(bRow + k0 + sb);          // global_load_b128
      b.h[1] = *(const v8h*)(bRow + k0 + sb + 16);
      acc = __builtin_amdgcn_wmma_f32_16x16x32_f16(
          false, a.v, false, b.v, (short)0, acc, false, false);
    }
    const int n = n0 + l15;
    const float bv = bias[n];
#pragma unroll
    for (int r = 0; r < 8; ++r) {
      int m = r + (half << 3);
      float v = acc[r] + bv;
      if (relu) v = fmaxf(v, 0.0f);
      sOut[(row0 + m) * Ndim + n] = (_Float16)v;
    }
  }
}

// ---------------- fused token kernel: gather + 4-layer MLP + max-pool -------
// 4 tokens per block => M = 64 neighbor rows; each weight byte fetched from
// L2 once per 4 tokens (4x less weight traffic than 1 token/block).
__global__ __launch_bounds__(1024) void token_kernel(
    const float* __restrict__ features, const int* __restrict__ knnIdx,
    const _Float16* __restrict__ w1t, const float* __restrict__ b1,
    const _Float16* __restrict__ w2t, const float* __restrict__ b2,
    const _Float16* __restrict__ w3t, const float* __restrict__ b3,
    const _Float16* __restrict__ w4t, const float* __restrict__ b4,
    _Float16* __restrict__ pooled) {
  __shared__ __align__(16) _Float16 sAct0[TPB_TOK * 16 * TD];   // 96 KB
  __shared__ __align__(16) _Float16 sAct1[TPB_TOK * 16 * TD];   // 96 KB
  __shared__ int sKnn[TPB_TOK * KNN];
  const int tid  = threadIdx.x;
  const int tok0 = blockIdx.x * TPB_TOK;     // 128 % TPB_TOK == 0 -> same event
  const int bev  = tok0 >> 7;
  if (tid < TPB_TOK * KNN) sKnn[tid] = knnIdx[tok0 * KNN + tid];
  __syncthreads();
  // gather neighbor features into padded f16 [64][32] (K padded 6 -> 32)
  for (int e = tid; e < TPB_TOK * 16 * 32; e += 1024) {
    int k = e >> 5, f = e & 31;
    float v = (f < FEAT)
        ? features[((size_t)(bev * P_PT + sKnn[k])) * FEAT + f] : 0.0f;
    sAct0[k * 32 + f] = (_Float16)v;
  }
  __syncthreads();
  const int wave = tid >> 5, lane = tid & 31;
  const int mt = wave & 3;        // row-tile (token) 0..3
  const int ng = wave >> 2;       // N-group 0..7
  mlp_layer4(sAct0,  32, w1t, b1, sAct1, 256, true,  mt, ng, 8, lane);
  __syncthreads();
  mlp_layer4(sAct1, 256, w2t, b2, sAct0, 512, true,  mt, ng, 8, lane);
  __syncthreads();
  mlp_layer4(sAct0, 512, w3t, b3, sAct1, 768, true,  mt, ng, 8, lane);
  __syncthreads();
  mlp_layer4(sAct1, 768, w4t, b4, sAct0, 768, false, mt, ng, 8, lane);
  __syncthreads();
  // max-pool over the 16 neighbors of each of the 4 tokens
  for (int e = tid; e < TPB_TOK * TD; e += 1024) {
    int t = e / TD, d = e - t * TD;
    float mx = -3.0e38f;
#pragma unroll
    for (int k = 0; k < KNN; ++k)
      mx = fmaxf(mx, (float)sAct0[(t * 16 + k) * TD + d]);
    pooled[(size_t)(tok0 + t) * TD + d] = (_Float16)mx;
  }
}

// ---------------- attention MLP GEMM: [2048,768] @ Wt[768,768] -------------
__global__ __launch_bounds__(256) void gemm_tok(const _Float16* __restrict__ A,
                                                const _Float16* __restrict__ Wt,
                                                const float* __restrict__ bias,
                                                _Float16* __restrict__ outH,
                                                float* __restrict__ outF,
                                                int relu) {
  const int tid  = threadIdx.x;
  const int wave = tid >> 5, lane = tid & 31;
  const int half = lane >> 4, l15 = lane & 15;
  const int sa = half ? 8 : 0, sb = half ? 8 : 0;
  const int row0 = blockIdx.x * 16;
  const _Float16* aRow = A + (size_t)(row0 + l15) * TD;
  for (int n0 = wave * 16; n0 < TD; n0 += 128) {
    const _Float16* bRow = Wt + (size_t)(n0 + l15) * TD;
    v8f acc = {};
    for (int k0 = 0; k0 < TD; k0 += 32) {
      union { v16h v; v8h h[2]; } a, b;
      a.h[0] = *(const v8h*)(aRow + k0 + sa);
      a.h[1] = *(const v8h*)(aRow + k0 + sa + 16);
      b.h[0] = *(const v8h*)(bRow + k0 + sb);
      b.h[1] = *(const v8h*)(bRow + k0 + sb + 16);
      acc = __builtin_amdgcn_wmma_f32_16x16x32_f16(
          false, a.v, false, b.v, (short)0, acc, false, false);
    }
    const int n = n0 + l15;
    const float bv = bias[n];
#pragma unroll
    for (int r = 0; r < 8; ++r) {
      int m = row0 + r + (half << 3);
      float v = acc[r] + bv;
      if (relu) v = fmaxf(v, 0.0f);
      if (outH) outH[(size_t)m * TD + n] = (_Float16)v;
      else      outF[(size_t)m * TD + n] = v;
    }
  }
}

// ---------------- masks = 1.0f ----------------------------------------------
__global__ void fill_ones_kernel(float* __restrict__ out, int n) {
  int id = blockIdx.x * 256 + threadIdx.x;
  if (id < n) out[id] = 1.0f;
}

// ---------------------------------------------------------------------------
extern "C" void kernel_launch(void* const* d_in, const int* in_sizes, int n_in,
                              void* d_out, int out_size, void* d_ws, size_t ws_size,
                              hipStream_t stream) {
  (void)in_sizes; (void)n_in; (void)out_size; (void)ws_size;
  const float* coordinates = (const float*)d_in[0];
  const float* features    = (const float*)d_in[1];
  const float* W1 = (const float*)d_in[2];   const float* b1 = (const float*)d_in[3];
  const float* W2 = (const float*)d_in[4];   const float* b2 = (const float*)d_in[5];
  const float* W3 = (const float*)d_in[6];   const float* b3 = (const float*)d_in[7];
  const float* W4 = (const float*)d_in[8];   const float* b4 = (const float*)d_in[9];
  const float* Wa1 = (const float*)d_in[10]; const float* ba1 = (const float*)d_in[11];
  const float* Wa2 = (const float*)d_in[12]; const float* ba2 = (const float*)d_in[13];

  // output layout: tokens [16*128*768] | centroids [16*128*4] | masks [16*128]
  float* tokens   = (float*)d_out;
  float* centOut  = tokens + (size_t)B_EV * MTOK * TD;
  float* masksOut = centOut + (size_t)B_EV * MTOK * 4;

  // workspace layout (256-byte aligned regions, ~11 MB total)
  char* ws = (char*)d_ws;
  size_t off = 0;
  auto take = [&](size_t bytes) -> char* {
    char* p = ws + off;
    off = (off + bytes + 255) & ~(size_t)255;
    return p;
  };
  _Float16* w1t  = (_Float16*)take((size_t)256 * 32  * 2);
  _Float16* w2t  = (_Float16*)take((size_t)512 * 256 * 2);
  _Float16* w3t  = (_Float16*)take((size_t)768 * 512 * 2);
  _Float16* w4t  = (_Float16*)take((size_t)768 * 768 * 2);
  _Float16* wa1t = (_Float16*)take((size_t)768 * 768 * 2);
  _Float16* wa2t = (_Float16*)take((size_t)768 * 768 * 2);
  float*    centWs = (float*)take((size_t)B_EV * MTOK * 4 * 4);
  int*      fpsIdx = (int*)take((size_t)B_EV * MTOK * 4);
  int*      knnIdx = (int*)take((size_t)B_EV * MTOK * KNN * 4);
  _Float16* pooled = (_Float16*)take((size_t)B_EV * MTOK * TD * 2);
  _Float16* hidden = (_Float16*)take((size_t)B_EV * MTOK * TD * 2);

  // 1) weight transpose/convert (f32 [K,N] -> f16 [N,Kpad])
  auto launch_wcvt = [&](const float* W, _Float16* Wt, int K, int N, int Kpad) {
    int total = N * Kpad;
    wcvt_kernel<<<(total + 255) / 256, 256, 0, stream>>>(W, Wt, K, N, Kpad);
  };
  launch_wcvt(W1,  w1t,  FEAT, 256, 32);
  launch_wcvt(W2,  w2t,  256,  512, 256);
  launch_wcvt(W3,  w3t,  512,  768, 512);
  launch_wcvt(W4,  w4t,  768,  768, 768);
  launch_wcvt(Wa1, wa1t, 768,  768, 768);
  launch_wcvt(Wa2, wa2t, 768,  768, 768);

  // 2) farthest point sampling (writes centroids to d_out and workspace)
  fps_kernel<<<B_EV, 1024, 0, stream>>>(coordinates, centOut, centWs, fpsIdx);

  // 3) kNN per centroid
  knn_kernel<<<B_EV * MTOK, 256, 0, stream>>>(coordinates, centWs, knnIdx);

  // 4) fused gather + point-MLP (WMMA) + max-pool, 4 tokens per block
  token_kernel<<<(B_EV * MTOK) / TPB_TOK, 1024, 0, stream>>>(
      features, knnIdx, w1t, b1, w2t, b2, w3t, b3, w4t, b4, pooled);

  // 5) attention MLP (WMMA GEMMs)
  gemm_tok<<<(B_EV * MTOK) / 16, 256, 0, stream>>>(pooled, wa1t, ba1,
                                                   hidden, (float*)nullptr, 1);
  gemm_tok<<<(B_EV * MTOK) / 16, 256, 0, stream>>>(hidden, wa2t, ba2,
                                                   (_Float16*)nullptr, tokens, 0);

  // 6) masks (all valid since P > MAX_TOKENS)
  fill_ones_kernel<<<(B_EV * MTOK + 255) / 256, 256, 0, stream>>>(masksOut, B_EV * MTOK);
}
